// T5Attention_87033217286251
// MI455X (gfx1250) — matrix operand order repaired and checked by reading, not compiled, not run on hardware
//
#include <hip/hip_runtime.h>
#include <hip/hip_bf16.h>

// ---------------------------------------------------------------------------
// T5 encoder self-attention for MI455X (gfx1250), wave32 + WMMA bf16.
//   X[B,S,D] --(3x WMMA GEMM, LDS double-buffered weights)--> Q,K,V bf16
//   flash-attention (per (b,h), 64-q tiles, online softmax, T5 bias inline)
//   ctx bf16 [B*S, D] --(WMMA GEMM)--> out f32 [B,S,D]
// Async global->LDS staging (ASYNCcnt) used when the toolchain exposes the
// gfx1250 builtins; otherwise falls back to synchronous staging.
// ---------------------------------------------------------------------------

#define DEVFN __device__ __forceinline__

typedef unsigned short u16t;
typedef __attribute__((ext_vector_type(16))) __bf16 v16bf;
typedef __attribute__((ext_vector_type(8)))  float  v8f;

constexpr int Bc  = 2;
constexpr int Sc  = 2048;
constexpr int Dc  = 1024;
constexpr int Hc  = 16;
constexpr int DKc = 64;
constexpr int PAD = 80;   // LDS row stride (u16 units): 160B -> 16B-aligned rows

#if defined(__has_builtin)
#if __has_builtin(__builtin_amdgcn_global_load_async_to_lds_b128) && \
    __has_builtin(__builtin_amdgcn_s_wait_asynccnt)
#define HAVE_ASYNC_LDS 1
#endif
#endif
#ifndef HAVE_ASYNC_LDS
#define HAVE_ASYNC_LDS 0
#endif

union Frag {
  v16bf v;
  uint4 q[2];
  u16t  u[16];
};

// Builtin parameter types per hipcc diagnostic: int4* in AS1 (global) and the
// LDS destination in AS3.
typedef int v4i_ld __attribute__((vector_size(4 * sizeof(int))));
typedef __attribute__((address_space(1))) v4i_ld* gptr_b128;
typedef __attribute__((address_space(3))) v4i_ld* lptr_b128;

// 16-byte global -> LDS copy; async (ASYNCcnt) when available.
DEVFN void lds_copy16(const u16t* g, u16t* l) {
#if HAVE_ASYNC_LDS
  __builtin_amdgcn_global_load_async_to_lds_b128((gptr_b128)g, (lptr_b128)l, 0, 0);
#else
  *(uint4*)l = *(const uint4*)g;
#endif
}

DEVFN void async_wait0() {
#if HAVE_ASYNC_LDS
  __builtin_amdgcn_s_wait_asynccnt(0);
#endif
}

DEVFN u16t f2b(float f) {
  union { float f; unsigned u; } x; x.f = f;
  unsigned r = x.u + 0x7fffu + ((x.u >> 16) & 1u);   // round-to-nearest-even
  return (u16t)(r >> 16);
}

// T5 bidirectional relative-position bucket bias. rp = k - q.
DEVFN float t5_bias(int rp, const float* __restrict__ rb, int h) {
  int bucket = (rp > 0) ? 16 : 0;
  int a = (rp < 0) ? -rp : rp;
  if (a < 8) {
    bucket += a;
  } else {
    // 8 + log(a/8)/log(16) * 8, clamped to 15
    int li = 8 + (int)(__logf((float)a * 0.125f) * (8.0f / 2.772588722239781f));
    bucket += (li < 15) ? li : 15;
  }
  return rb[bucket * Hc + h];
}

__global__ void f32_to_bf16(const float* __restrict__ in, u16t* __restrict__ out, int n) {
  int i = blockIdx.x * blockDim.x + threadIdx.x;
  if (i < n) out[i] = f2b(in[i]);
}

// ---------------------------------------------------------------------------
// C = A[M,K] * W[N,K]^T (torch Linear). 4 waves/block; each wave: 16x64 out.
// W tile (64n x 64k) staged in LDS, double-buffered via async-to-LDS; all 4
// waves share it. A rows streamed from global (L2-resident) with prefetch.
// MODE 0: write f32 row-major. MODE 1: write bf16 row-major.
// ---------------------------------------------------------------------------
template <int MODE>
__global__ __launch_bounds__(128)
void wmma_gemm_bf16(const u16t* __restrict__ A, const u16t* __restrict__ W,
                    float* __restrict__ outF, u16t* __restrict__ outB,
                    int M, int N, int K) {
  __shared__ u16t sW[2][64][PAD];

  const int tid  = threadIdx.x;
  const int lane = tid & 31;
  const int wave = tid >> 5;
  const int ln   = lane & 15;
  const int hi   = lane >> 4;
  const int m0   = blockIdx.y * 64 + wave * 16;
  const int n0   = blockIdx.x * 64;

  const u16t* arow = A + (size_t)(m0 + ln) * K;

  // stage W[n0..n0+63][kc..kc+63] -> sW[buf] (coalesced 16B copies)
  auto stageW = [&](int kc, int buf) {
    for (int i = tid; i < 512; i += 128) {
      int r = i >> 3, c8 = (i & 7) * 8;
      lds_copy16(W + (size_t)(n0 + r) * K + kc + c8, &sW[buf][r][c8]);
    }
  };

  stageW(0, 0);

  v8f acc[4] = {};
  for (int kc = 0; kc < K; kc += 64) {
    const int buf = (kc >> 6) & 1;
    async_wait0();       // current buffer's async copies complete
    __syncthreads();     // ...and visible to all waves
    if (kc + 64 < K) {
      stageW(kc + 64, buf ^ 1);                  // overlap next chunk with WMMA
      __builtin_prefetch(arow + kc + 64, 0, 0);  // global_prefetch for A stream
    }

#pragma unroll
    for (int kk = 0; kk < 2; ++kk) {
      Frag a;
      const u16t* ap = arow + kc + kk * 32 + hi * 8;
      a.q[0] = *(const uint4*)(ap);
      a.q[1] = *(const uint4*)(ap + 16);
#pragma unroll
      for (int ns = 0; ns < 4; ++ns) {
        Frag bw;
        const u16t* bp = &sW[buf][ns * 16 + ln][kk * 32 + hi * 8];
        bw.q[0] = *(const uint4*)(bp);
        bw.q[1] = *(const uint4*)(bp + 16);
        acc[ns] = __builtin_amdgcn_wmma_f32_16x16x32_bf16(
            false, a.v, false, bw.v, (short)0, acc[ns], false, false);
      }
    }
  }

#pragma unroll
  for (int ns = 0; ns < 4; ++ns) {
#pragma unroll
    for (int i = 0; i < 8; ++i) {
      const int row = m0 + i + hi * 8;     // C layout: lanes 16-31 hold M+8
      const int col = n0 + ns * 16 + ln;
      if (MODE == 0) outF[(size_t)row * N + col] = acc[ns][i];
      else           outB[(size_t)row * N + col] = f2b(acc[ns][i]);
    }
  }
}

// ---------------------------------------------------------------------------
// Fused flash attention with T5 relative bias.
// grid = (S/64, B*H), block = 128 (4 waves). Each wave: 16 query rows.
// ---------------------------------------------------------------------------
__global__ __launch_bounds__(128)
void t5_flash_attn(const u16t* __restrict__ Qb, const u16t* __restrict__ Kb,
                   const u16t* __restrict__ Vb, const float* __restrict__ rel_bias,
                   u16t* __restrict__ Ctx) {
  __shared__ u16t sQ[64][PAD];        // [q][d]
  __shared__ u16t sK[64][PAD];        // [k][d]
  __shared__ u16t sVT[64][PAD];       // [d][k]  (transposed for B-fragments)
  __shared__ u16t sP[4][16][PAD];     // per-wave P tile [q][k]

  const int tid  = threadIdx.x;
  const int lane = tid & 31, wave = tid >> 5;
  const int ln   = lane & 15, hi = lane >> 4;
  const int bh   = blockIdx.y, b = bh / Hc, h = bh % Hc;
  const int q0   = blockIdx.x * 64;
  const size_t headoff = (size_t)h * DKc;

  // stage Q tile once (async; first tile's wait covers it)
  for (int i = tid; i < 512; i += 128) {
    int r = i >> 3, c8 = (i & 7) * 8;
    lds_copy16(Qb + (size_t)(b * Sc + q0 + r) * Dc + headoff + c8, &sQ[r][c8]);
  }

  float mrun[8], lrun[8];
  v8f o[4] = {};
#pragma unroll
  for (int i = 0; i < 8; ++i) { mrun[i] = -3.0e38f; lrun[i] = 0.0f; }

  for (int kt0 = 0; kt0 < Sc; kt0 += 64) {
    __syncthreads();
    // stage K (async) and V^T (sync transpose-scatter) tiles
    for (int i = tid; i < 512; i += 128) {
      int r = i >> 3, c8 = (i & 7) * 8;
      lds_copy16(Kb + (size_t)(b * Sc + kt0 + r) * Dc + headoff + c8, &sK[r][c8]);
      uint4 vv = *(const uint4*)(Vb + (size_t)(b * Sc + kt0 + r) * Dc + headoff + c8);
      const u16t* e = (const u16t*)&vv;
#pragma unroll
      for (int j = 0; j < 8; ++j) sVT[c8 + j][r] = e[j];
    }
    async_wait0();
    __syncthreads();

    // ---- scores: c[kt] = Qtile(16xDK) . Ktile(16xDK)^T, kt = 4 key subtiles
    v8f c[4] = {};
#pragma unroll
    for (int d0 = 0; d0 < DKc; d0 += 32) {
      Frag a;
      const u16t* qp = &sQ[wave * 16 + ln][d0 + hi * 8];
      a.q[0] = *(const uint4*)(qp);
      a.q[1] = *(const uint4*)(qp + 16);
#pragma unroll
      for (int kt = 0; kt < 4; ++kt) {
        Frag bk;
        const u16t* kp = &sK[kt * 16 + ln][d0 + hi * 8];
        bk.q[0] = *(const uint4*)(kp);
        bk.q[1] = *(const uint4*)(kp + 16);
        c[kt] = __builtin_amdgcn_wmma_f32_16x16x32_bf16(
            false, a.v, false, bk.v, (short)0, c[kt], false, false);
      }
    }

    // ---- add T5 relative-position bias
    const int qbase = q0 + wave * 16 + hi * 8;
#pragma unroll
    for (int kt = 0; kt < 4; ++kt) {
      const int kg = kt0 + kt * 16 + ln;
#pragma unroll
      for (int i = 0; i < 8; ++i)
        c[kt][i] += t5_bias(kg - (qbase + i), rel_bias, h);
    }

    // ---- online softmax (row stats per fragment-row; 16-lane reductions)
#pragma unroll
    for (int i = 0; i < 8; ++i) {
      float t = fmaxf(fmaxf(c[0][i], c[1][i]), fmaxf(c[2][i], c[3][i]));
#pragma unroll
      for (int off = 1; off < 16; off <<= 1)
        t = fmaxf(t, __shfl_xor(t, off, 32));
      const float mn = fmaxf(mrun[i], t);
      const float sc = __expf(mrun[i] - mn);
      mrun[i] = mn;
      lrun[i] *= sc;
#pragma unroll
      for (int dch = 0; dch < 4; ++dch) o[dch][i] *= sc;
      float s = 0.0f;
#pragma unroll
      for (int kt = 0; kt < 4; ++kt) {
        c[kt][i] = __expf(c[kt][i] - mn);
        s += c[kt][i];
      }
#pragma unroll
      for (int off = 1; off < 16; off <<= 1)
        s += __shfl_xor(s, off, 32);
      lrun[i] += s;
    }

    // ---- P (C-fragment layout) -> LDS -> A-fragment layout
#pragma unroll
    for (int kt = 0; kt < 4; ++kt)
#pragma unroll
      for (int i = 0; i < 8; ++i)
        sP[wave][i + 8 * hi][kt * 16 + ln] = f2b(c[kt][i]);

    __builtin_amdgcn_wave_barrier();  // wave-private LDS; DS ops in-order per wave

    // ---- O += P(16x64) . V(64xDK)
#pragma unroll
    for (int kk = 0; kk < 2; ++kk) {
      Frag pa;
      const u16t* pp = &sP[wave][ln][kk * 32 + hi * 8];
      pa.q[0] = *(const uint4*)(pp);
      pa.q[1] = *(const uint4*)(pp + 16);
#pragma unroll
      for (int dch = 0; dch < 4; ++dch) {
        Frag bv;
        const u16t* vp = &sVT[dch * 16 + ln][kk * 32 + hi * 8];
        bv.q[0] = *(const uint4*)(vp);
        bv.q[1] = *(const uint4*)(vp + 16);
        o[dch] = __builtin_amdgcn_wmma_f32_16x16x32_bf16(
            false, pa.v, false, bv.v, (short)0, o[dch], false, false);
      }
    }
  }

  // ---- epilogue: O / l -> ctx bf16 [B*S, D]
#pragma unroll
  for (int dch = 0; dch < 4; ++dch)
#pragma unroll
    for (int i = 0; i < 8; ++i) {
      const int q = q0 + wave * 16 + i + 8 * hi;
      const float val = o[dch][i] / lrun[i];
      Ctx[(size_t)(b * Sc + q) * Dc + headoff + dch * 16 + ln] = f2b(val);
    }
}

// ---------------------------------------------------------------------------
extern "C" void kernel_launch(void* const* d_in, const int* in_sizes, int n_in,
                              void* d_out, int out_size, void* d_ws, size_t ws_size,
                              hipStream_t stream) {
  (void)in_sizes; (void)n_in; (void)out_size; (void)ws_size;

  const float* X  = (const float*)d_in[0];
  const float* Wq = (const float*)d_in[1];
  const float* Wk = (const float*)d_in[2];
  const float* Wv = (const float*)d_in[3];
  const float* Wo = (const float*)d_in[4];
  const float* rb = (const float*)d_in[5];
  float* out = (float*)d_out;

  constexpr size_t NX = (size_t)Bc * Sc * Dc;  // 4,194,304
  constexpr size_t NW = (size_t)Dc * Dc;       // 1,048,576

  // workspace layout (u16 bf16 elements): ~50.3 MB total
  u16t* w   = (u16t*)d_ws;
  u16t* Xb  = w;
  u16t* Wqb = Xb + NX;
  u16t* Wkb = Wqb + NW;
  u16t* Wvb = Wkb + NW;
  u16t* Wob = Wvb + NW;
  u16t* Qb  = Wob + NW;
  u16t* Kb  = Qb + NX;
  u16t* Vbf = Kb + NX;
  u16t* Ctx = Vbf + NX;

  f32_to_bf16<<<(unsigned)((NX + 255) / 256), 256, 0, stream>>>(X, Xb, (int)NX);
  f32_to_bf16<<<(unsigned)((NW + 255) / 256), 256, 0, stream>>>(Wq, Wqb, (int)NW);
  f32_to_bf16<<<(unsigned)((NW + 255) / 256), 256, 0, stream>>>(Wk, Wkb, (int)NW);
  f32_to_bf16<<<(unsigned)((NW + 255) / 256), 256, 0, stream>>>(Wv, Wvb, (int)NW);
  f32_to_bf16<<<(unsigned)((NW + 255) / 256), 256, 0, stream>>>(Wo, Wob, (int)NW);

  dim3 gg(Dc / 64, (Bc * Sc) / 64);  // (16, 64)
  wmma_gemm_bf16<1><<<gg, 128, 0, stream>>>(Xb, Wqb, nullptr, Qb,  Bc * Sc, Dc, Dc);
  wmma_gemm_bf16<1><<<gg, 128, 0, stream>>>(Xb, Wkb, nullptr, Kb,  Bc * Sc, Dc, Dc);
  wmma_gemm_bf16<1><<<gg, 128, 0, stream>>>(Xb, Wvb, nullptr, Vbf, Bc * Sc, Dc, Dc);

  dim3 ga(Sc / 64, Bc * Hc);  // (32, 32)
  t5_flash_attn<<<ga, 128, 0, stream>>>(Qb, Kb, Vbf, rb, Ctx);

  wmma_gemm_bf16<0><<<gg, 128, 0, stream>>>(Ctx, Wob, out, nullptr, Bc * Sc, Dc, Dc);
}
